// ExperimentalGNN_4166118277632
// MI455X (gfx1250) — matrix operand order repaired and checked by reading, not compiled, not run on hardware
//
#include <hip/hip_runtime.h>
#include <hip/hip_bf16.h>

// ---------------------------------------------------------------------------
// Types for CDNA5 WMMA (wave32, 16x16x32 f16 -> f32)
// ---------------------------------------------------------------------------
typedef __attribute__((__ext_vector_type__(16))) _Float16 v16h;
typedef __attribute__((__ext_vector_type__(8)))  float    v8f;

union HV { v16h v; _Float16 e[16]; uint4 q[2]; };
union FV { v8f  v; float    e[8];  };

constexpr unsigned ORD_NEG = 0x00800000u; // f2ord(-FLT_MAX)

__device__ __forceinline__ unsigned f2ord(float f) {
  unsigned u = __float_as_uint(f);
  return (u & 0x80000000u) ? ~u : (u | 0x80000000u);
}
__device__ __forceinline__ float ord2f(unsigned u) {
  return (u & 0x80000000u) ? __uint_as_float(u & 0x7fffffffu) : __uint_as_float(~u);
}
__device__ __forceinline__ float sigmoidf_(float x) { return 1.f / (1.f + expf(-x)); }

// ---------------------------------------------------------------------------
// Fragment-order LDS staging.
//   A tile: 128 rows x 64 k, stored so that wave w / k-chunk kc / lane has its
//           16 halves contiguous:  offset = ((w*2+kc)*32 + lane)*16 + idx
//           element = X[w*16 + (lane&15)][kc*32 + (idx>>3)*16 + (lane>>4)*8 + (idx&7)]
//   B tile: 64x64 weights, offset = ((kc*4+nt)*32 + lane)*16 + idx
//           element = W[kc*32 + (idx>>3)*16 + (lane>>4)*8 + (idx&7)][nt*16 + (lane&15)]
// Each fragment load is then two aligned ds_load_b128.
// ---------------------------------------------------------------------------
__device__ __forceinline__ v16h ld_frag(const _Float16* p) {
  HV u;
  u.q[0] = *(const uint4*)p;
  u.q[1] = *(const uint4*)(p + 8);
  return u.v;
}

__device__ __forceinline__ void stage_weights_frag(const float* __restrict__ Wp,
                                                   _Float16* sBw, int tid) {
  for (int t = tid; t < 4096; t += 256) {
    int idx = t & 15, lane = (t >> 4) & 31, nt = (t >> 9) & 3, kc = t >> 11;
    int k = kc * 32 + ((idx >> 3) << 4) + ((lane >> 4) << 3) + (idx & 7);
    int n = nt * 16 + (lane & 15);
    sBw[t] = (_Float16)Wp[k * 64 + n];
  }
}

template <bool ADD2>
__device__ __forceinline__ void stage_acts_frag(const float* __restrict__ X,
                                                const float* __restrict__ X2,
                                                _Float16* sA, int base, int R, int tid) {
  // prefetch next block's tile while we stage this one (global_prefetch_b8)
  {
    int prow = base + 128 + (tid >> 1);
    if (prow < R) __builtin_prefetch(&X[prow * 64 + (tid & 1) * 32], 0, 0);
  }
  for (int t = tid; t < 8192; t += 256) {
    int idx = t & 15, lane = (t >> 4) & 31, kc = (t >> 9) & 1, w = t >> 10;
    int row = base + w * 16 + (lane & 15);
    int k = kc * 32 + ((idx >> 3) << 4) + ((lane >> 4) << 3) + (idx & 7);
    float v = 0.f;
    if (row < R) {
      v = X[row * 64 + k];
      if (ADD2) v += X2[row * 64 + k];
    }
    sA[t] = (_Float16)v;
  }
}

// One wave: 16x64 tile of X(16x64) @ W(64x64); acc pre-loaded with bias.
__device__ __forceinline__ void wmma_core(const _Float16* sA, const _Float16* sBw,
                                          int wave, int lane, v8f acc[4]) {
#pragma unroll
  for (int kc = 0; kc < 2; ++kc) {
    v16h a = ld_frag(sA + (((wave * 2 + kc) * 32) + lane) * 16);
#pragma unroll
    for (int nt = 0; nt < 4; ++nt) {
      v16h b = ld_frag(sBw + (((kc * 4 + nt) * 32) + lane) * 16);
      acc[nt] = __builtin_amdgcn_wmma_f32_16x16x32_f16(
          false, a, false, b, (short)0, acc[nt], false, false);
    }
  }
}

// ---------------------------------------------------------------------------
// Fused 64x64 linear over R rows.  EPI: 0=none, 1=LN+SiLU, 2=LN+residual-add
// ---------------------------------------------------------------------------
template <int EPI, bool ADD2>
__global__ __launch_bounds__(256) void gemm64_fused(
    const float* __restrict__ X, const float* __restrict__ X2,
    const float* __restrict__ Wp, const float* __restrict__ bias,
    const float* __restrict__ gln, const float* __restrict__ bln,
    const float* __restrict__ resid, float* __restrict__ Y, int R) {
  __shared__ __align__(16) _Float16 sA[8192];
  __shared__ __align__(16) _Float16 sBw[4096];

  const int tid = threadIdx.x;
  const int base = blockIdx.x * 128;

  stage_weights_frag(Wp, sBw, tid);
  stage_acts_frag<ADD2>(X, X2, sA, base, R, tid);
  __syncthreads();

  const int lane = tid & 31, wave = tid >> 5;
  const int n15 = lane & 15, hi = lane >> 4;

  v8f acc[4];
#pragma unroll
  for (int nt = 0; nt < 4; ++nt) {
    float bv = bias[nt * 16 + n15];          // bias folded into accumulator init
    FV zi;
#pragma unroll
    for (int j = 0; j < 8; ++j) zi.e[j] = bv;
    acc[nt] = zi.v;
  }

  wmma_core(sA, sBw, wave, lane, acc);

  float z[4][8];
#pragma unroll
  for (int nt = 0; nt < 4; ++nt) {
    FV cu; cu.v = acc[nt];
#pragma unroll
    for (int r = 0; r < 8; ++r) z[nt][r] = cu.e[r];
  }

#pragma unroll
  for (int r = 0; r < 8; ++r) {
    int row = base + wave * 16 + hi * 8 + r;
    if (EPI == 0) {
      if (row < R) {
#pragma unroll
        for (int nt = 0; nt < 4; ++nt) Y[row * 64 + nt * 16 + n15] = z[nt][r];
      }
    } else {
      float s = z[0][r] + z[1][r] + z[2][r] + z[3][r];
      float q = z[0][r]*z[0][r] + z[1][r]*z[1][r] + z[2][r]*z[2][r] + z[3][r]*z[3][r];
#pragma unroll
      for (int off = 1; off < 16; off <<= 1) {
        s += __shfl_xor(s, off, 16);
        q += __shfl_xor(q, off, 16);
      }
      float mean = s * (1.f / 64.f);
      float var  = q * (1.f / 64.f) - mean * mean;
      float rstd = rsqrtf(var + 1e-5f);
      if (row < R) {
#pragma unroll
        for (int nt = 0; nt < 4; ++nt) {
          int col = nt * 16 + n15;
          float y = (z[nt][r] - mean) * rstd * gln[col] + bln[col];
          if (EPI == 1) y = y * sigmoidf_(y);               // SiLU
          else          y = resid[row * 64 + col] + y;       // residual
          Y[row * 64 + col] = y;
        }
      }
    }
  }
}

// ---------------------------------------------------------------------------
// GINE message:  agg[dst] += relu(h[src] + e @ W_edge + b)   (WMMA + scatter)
// ---------------------------------------------------------------------------
__global__ __launch_bounds__(256) void gine_msg(
    const float* __restrict__ E_, const float* __restrict__ Wp,
    const float* __restrict__ bias, const float* __restrict__ h,
    const int* __restrict__ src, const int* __restrict__ dst,
    float* __restrict__ agg, int R) {
  __shared__ __align__(16) _Float16 sA[8192];
  __shared__ __align__(16) _Float16 sBw[4096];

  const int tid = threadIdx.x;
  const int base = blockIdx.x * 128;

  stage_weights_frag(Wp, sBw, tid);
  stage_acts_frag<false>(E_, nullptr, sA, base, R, tid);
  __syncthreads();

  const int lane = tid & 31, wave = tid >> 5;
  const int n15 = lane & 15, hi = lane >> 4;

  v8f acc[4];
#pragma unroll
  for (int nt = 0; nt < 4; ++nt) {
    float bv = bias[nt * 16 + n15];
    FV zi;
#pragma unroll
    for (int j = 0; j < 8; ++j) zi.e[j] = bv;
    acc[nt] = zi.v;
  }

  wmma_core(sA, sBw, wave, lane, acc);

  FV cu[4];
#pragma unroll
  for (int nt = 0; nt < 4; ++nt) cu[nt].v = acc[nt];

#pragma unroll
  for (int r = 0; r < 8; ++r) {
    int row = base + wave * 16 + hi * 8 + r;
    if (row < R) {
      int sN = src[row], dN = dst[row];
#pragma unroll
      for (int nt = 0; nt < 4; ++nt) {
        int col = nt * 16 + n15;
        float v = cu[nt].e[r] + h[sN * 64 + col];
        v = fmaxf(v, 0.f);
        atomicAdd(&agg[dN * 64 + col], v);
      }
    }
  }
}

// ---------------------------------------------------------------------------
// Transformer attention over edges (8 heads, C=8)
// ---------------------------------------------------------------------------
__global__ void attn_alpha(const float* __restrict__ q, const float* __restrict__ k,
                           const float* __restrict__ eh, const int* __restrict__ src,
                           const int* __restrict__ dst, float* __restrict__ alpha,
                           unsigned* __restrict__ mord, int E) {
  int t = blockIdx.x * blockDim.x + threadIdx.x;
  if (t >= E * 8) return;
  int e = t >> 3, hd = t & 7;
  int s = src[e], d = dst[e];
  float acc = 0.f;
#pragma unroll
  for (int c = 0; c < 8; ++c)
    acc += q[d * 64 + hd * 8 + c] * (k[s * 64 + hd * 8 + c] + eh[e * 64 + hd * 8 + c]);
  acc *= 0.35355339059327373f;  // 1/sqrt(8)
  alpha[t] = acc;
  atomicMax(&mord[d * 8 + hd], f2ord(acc));
}

__global__ void attn_exp(const float* __restrict__ alpha, const unsigned* __restrict__ mord,
                         const int* __restrict__ dst, float* __restrict__ exb,
                         float* __restrict__ sden, int E) {
  int t = blockIdx.x * blockDim.x + threadIdx.x;
  if (t >= E * 8) return;
  int e = t >> 3, hd = t & 7;
  int d = dst[e];
  float ex = expf(alpha[t] - ord2f(mord[d * 8 + hd]));
  exb[t] = ex;
  atomicAdd(&sden[d * 8 + hd], ex);
}

__global__ void attn_agg(const float* __restrict__ v, const float* __restrict__ eh,
                         const float* __restrict__ exb, const float* __restrict__ sden,
                         const int* __restrict__ src, const int* __restrict__ dst,
                         float* __restrict__ out, int E) {
  int t = blockIdx.x * blockDim.x + threadIdx.x;
  if (t >= E * 8) return;
  int e = t >> 3, hd = t & 7;
  int s = src[e], d = dst[e];
  float a = exb[t] / (sden[d * 8 + hd] + 1e-16f);
#pragma unroll
  for (int c = 0; c < 8; ++c)
    atomicAdd(&out[d * 64 + hd * 8 + c],
              a * (v[s * 64 + hd * 8 + c] + eh[e * 64 + hd * 8 + c]));
}

// beta-gated combine + LayerNorm + residual, one wave per node
__global__ __launch_bounds__(256) void trans_combine(
    const float* __restrict__ outb, const float* __restrict__ xr,
    const float* __restrict__ betaW, const float* __restrict__ gln,
    const float* __restrict__ bln, float* __restrict__ h, int N) {
  int wave = blockIdx.x * 8 + (threadIdx.x >> 5);
  int lane = threadIdx.x & 31;
  if (wave >= N) return;
  float o0 = outb[wave * 64 + lane],      o1 = outb[wave * 64 + 32 + lane];
  float x0 = xr[wave * 64 + lane],        x1 = xr[wave * 64 + 32 + lane];
  float p = o0 * betaW[lane] + o1 * betaW[32 + lane]
          + x0 * betaW[64 + lane] + x1 * betaW[96 + lane]
          + (o0 - x0) * betaW[128 + lane] + (o1 - x1) * betaW[160 + lane];
#pragma unroll
  for (int off = 1; off < 32; off <<= 1) p += __shfl_xor(p, off, 32);
  float beta = sigmoidf_(p);
  float n0 = beta * x0 + (1.f - beta) * o0;
  float n1 = beta * x1 + (1.f - beta) * o1;
  float s = n0 + n1, q = n0 * n0 + n1 * n1;
#pragma unroll
  for (int off = 1; off < 32; off <<= 1) {
    s += __shfl_xor(s, off, 32);
    q += __shfl_xor(q, off, 32);
  }
  float mean = s * (1.f / 64.f), var = q * (1.f / 64.f) - mean * mean;
  float rstd = rsqrtf(var + 1e-5f);
  h[wave * 64 + lane]      += (n0 - mean) * rstd * gln[lane] + bln[lane];
  h[wave * 64 + 32 + lane] += (n1 - mean) * rstd * gln[32 + lane] + bln[32 + lane];
}

// ---------------------------------------------------------------------------
// Set2Set readout
// ---------------------------------------------------------------------------
__global__ __launch_bounds__(256) void lstm_step(
    float* __restrict__ qstar, float* __restrict__ hB, float* __restrict__ cB,
    const float* __restrict__ W_ih, const float* __restrict__ W_hh,
    const float* __restrict__ b_ih, const float* __restrict__ b_hh) {
  int b = blockIdx.x, t = threadIdx.x;
  __shared__ float z[256];
  float acc = b_ih[t] + b_hh[t];
  for (int k = 0; k < 128; ++k) acc += qstar[b * 128 + k] * W_ih[k * 256 + t];
  for (int k = 0; k < 64;  ++k) acc += hB[b * 64 + k]    * W_hh[k * 256 + t];
  z[t] = acc;
  __syncthreads();
  if (t < 64) {
    float ig = sigmoidf_(z[t]);
    float fg = sigmoidf_(z[64 + t]);
    float gg = tanhf(z[128 + t]);
    float og = sigmoidf_(z[192 + t]);
    float c = fg * cB[b * 64 + t] + ig * gg;
    float hh = og * tanhf(c);
    cB[b * 64 + t] = c;
    hB[b * 64 + t] = hh;
    qstar[b * 128 + t] = hh;        // first half of next q_star
    qstar[b * 128 + 64 + t] = 0.f;  // zero r-accumulator slot
  }
}

__global__ void s2s_energy(const float* __restrict__ x, const float* __restrict__ hB,
                           const int* __restrict__ batch, float* __restrict__ ener,
                           unsigned* __restrict__ mord, int N) {
  int n = blockIdx.x * blockDim.x + threadIdx.x;
  if (n >= N) return;
  int b = batch[n];
  float acc = 0.f;
  for (int j = 0; j < 64; ++j) acc += x[n * 64 + j] * hB[b * 64 + j];
  ener[n] = acc;
  atomicMax(&mord[b], f2ord(acc));
}

__global__ void s2s_expsum(const float* __restrict__ ener, const unsigned* __restrict__ mord,
                           const int* __restrict__ batch, float* __restrict__ exn,
                           float* __restrict__ sden, int N) {
  int n = blockIdx.x * blockDim.x + threadIdx.x;
  if (n >= N) return;
  int b = batch[n];
  float ex = expf(ener[n] - ord2f(mord[b]));
  exn[n] = ex;
  atomicAdd(&sden[b], ex);
}

__global__ void s2s_weighted(const float* __restrict__ x, const float* __restrict__ exn,
                             const float* __restrict__ sden, const int* __restrict__ batch,
                             float* __restrict__ qstar, int N) {
  int n = blockIdx.x * blockDim.x + threadIdx.x;
  if (n >= N) return;
  int b = batch[n];
  float a = exn[n] / (sden[b] + 1e-16f);
  for (int j = 0; j < 64; ++j)
    atomicAdd(&qstar[b * 128 + 64 + j], a * x[n * 64 + j]);
}

// ---------------------------------------------------------------------------
// Small dense kernels (head / encoders).  EPI: 0=none, 1=LN+SiLU
// ---------------------------------------------------------------------------
template <int EPI>
__global__ __launch_bounds__(256) void small_linear(
    const float* __restrict__ X1, int K1, const float* __restrict__ X2, int K2,
    const float* __restrict__ W, const float* __restrict__ bias,
    const float* __restrict__ gln, const float* __restrict__ bln,
    float* __restrict__ Y, int Nout) {
  int row = blockIdx.x, t = threadIdx.x;
  __shared__ float z[256];
  __shared__ float stats[2];
  float acc = 0.f;
  if (t < Nout) {
    acc = bias[t];
    for (int k = 0; k < K1; ++k) acc += X1[row * K1 + k] * W[k * Nout + t];
    if (X2)
      for (int k = 0; k < K2; ++k) acc += X2[row * K2 + k] * W[(K1 + k) * Nout + t];
    z[t] = acc;
  }
  if (EPI == 1) {
    __syncthreads();
    if (t == 0) {
      float s = 0.f, q = 0.f;
      for (int j = 0; j < Nout; ++j) { s += z[j]; q += z[j] * z[j]; }
      float mean = s / Nout;
      stats[0] = mean;
      stats[1] = rsqrtf(q / Nout - mean * mean + 1e-5f);
    }
    __syncthreads();
    if (t < Nout) {
      float y = (acc - stats[0]) * stats[1] * gln[t] + bln[t];
      Y[row * Nout + t] = y * sigmoidf_(y);
    }
  } else {
    if (t < Nout) Y[row * Nout + t] = acc;
  }
}

__global__ void gfx_kernel(const float* nA, const float* nB, const float* ss,
                           float* gfx, int B) {
  int b = blockIdx.x * blockDim.x + threadIdx.x;
  if (b >= B) return;
  float d = ss[b] + 1e-10f;
  gfx[b * 2 + 0] = nA[b] / d;
  gfx[b * 2 + 1] = nB[b] / d;
}

__global__ void final_out(const float* __restrict__ z2, const float* __restrict__ W,
                          const float* __restrict__ b, float* __restrict__ out, int B) {
  int r = blockIdx.x * blockDim.x + threadIdx.x;
  if (r >= B) return;
  float acc = b[0];
  for (int k = 0; k < 32; ++k) acc += z2[r * 32 + k] * W[k];
  out[r] = log1pf(expf(acc));  // softplus
}

// ---------------------------------------------------------------------------
// Utility fills / copies
// ---------------------------------------------------------------------------
__global__ void fill_f32(float* p, float v, int n) {
  int i = blockIdx.x * blockDim.x + threadIdx.x;
  if (i < n) p[i] = v;
}
__global__ void fill_u32(unsigned* p, unsigned v, int n) {
  int i = blockIdx.x * blockDim.x + threadIdx.x;
  if (i < n) p[i] = v;
}
__global__ void copy_strided(const float* src, int sstride, float* dst, int dstride,
                             int cols, int rows) {
  int i = blockIdx.x * blockDim.x + threadIdx.x;
  if (i >= cols * rows) return;
  int r = i / cols, c = i % cols;
  dst[r * dstride + c] = src[r * sstride + c];
}

// ---------------------------------------------------------------------------
// Host orchestration
// ---------------------------------------------------------------------------
extern "C" void kernel_launch(void* const* d_in, const int* in_sizes, int n_in,
                              void* d_out, int out_size, void* d_ws, size_t ws_size,
                              hipStream_t stream) {
  (void)in_sizes; (void)n_in; (void)out_size; (void)ws_size;
  const int N = 20000, NE = 200000, NB = 128;

  auto F = [&](int i) -> const float* { return (const float*)d_in[i]; };
  const float* x_in   = F(0);
  const int*   src    = (const int*)d_in[1];
  const int*   dst    = src + NE;
  const float* eattr  = F(2);
  const int*   batch  = (const int*)d_in[3];

  // ---- flattened param pytree indices (sorted dict keys, list order) ----
  int pi = 7;
  const int I_EDGE_ENC = pi; pi += 4;   // lin.W, lin.b, ln.b, ln.g
  const int I_FINAL    = pi; pi += 10;  // lin1.W,b lin2.W,b lin3.W,b ln1.b,g ln2.b,g
  const int I_GM       = pi; pi += 6;   // lin1.W,b lin2.W,b ln.b, ln.g
  int I_LAYER[10];
  for (int i = 0; i < 10; ++i) { I_LAYER[i] = pi; pi += (i % 2 == 0) ? 14 : 17; }
  const int I_NODE_ENC = pi; pi += 4;
  int I_POOL[5];
  for (int j = 0; j < 5; ++j) { I_POOL[j] = pi; pi += 4; }
  int I_RD[2];
  for (int j = 0; j < 2; ++j) { I_RD[j] = pi; pi += 4; }  // W_hh, W_ih, b_hh, b_ih
  const int I_RP = pi; pi += 4;

  // ---- workspace carve ----
  char* wsp = (char*)d_ws;
  auto alloc = [&](size_t nfl) -> float* {
    float* p = (float*)wsp;
    wsp += ((nfl * sizeof(float) + 255) / 256) * 256;
    return p;
  };
  float* h      = alloc((size_t)N * 64);
  float* e      = alloc((size_t)NE * 64);
  float* nbuf   = alloc((size_t)N * 64);
  float* ztmp   = alloc((size_t)N * 64);
  float* qb     = alloc((size_t)N * 64);
  float* kb     = alloc((size_t)N * 64);
  float* vb     = alloc((size_t)N * 64);
  float* skipb  = alloc((size_t)N * 64);
  float* ehb    = alloc((size_t)NE * 64);
  float* alphab = alloc((size_t)NE * 8);
  float* exb    = alloc((size_t)NE * 8);
  unsigned* mord  = (unsigned*)alloc((size_t)N * 8);
  float* sden   = alloc((size_t)N * 8);
  float* hB     = alloc((size_t)NB * 64);
  float* cB     = alloc((size_t)NB * 64);
  float* qstar  = alloc((size_t)NB * 128);
  float* ener   = alloc((size_t)N);
  float* exn    = alloc((size_t)N);
  unsigned* mordB = (unsigned*)alloc((size_t)NB);
  float* sdenB  = alloc((size_t)NB);
  float* r4     = alloc((size_t)NB * 256);
  float* rproj  = alloc((size_t)NB * 128);
  float* gfxb   = alloc((size_t)NB * 2);
  float* t1     = alloc((size_t)NB * 64);
  float* gf     = alloc((size_t)NB * 64);
  float* z1     = alloc((size_t)NB * 64);
  float* z2     = alloc((size_t)NB * 32);

  const int gN = (N + 127) / 128;
  const int gE = (NE + 127) / 128;
  const int gEH = (NE * 8 + 255) / 256;

  // ---- encoders ----
  small_linear<1><<<N, 64, 0, stream>>>(x_in, 4, nullptr, 0,
      F(I_NODE_ENC + 0), F(I_NODE_ENC + 1), F(I_NODE_ENC + 3), F(I_NODE_ENC + 2),
      h, 64);
  small_linear<1><<<NE, 64, 0, stream>>>(eattr, 3, nullptr, 0,
      F(I_EDGE_ENC + 0), F(I_EDGE_ENC + 1), F(I_EDGE_ENC + 3), F(I_EDGE_ENC + 2),
      e, 64);

  // ---- 10 message-passing layers ----
  for (int i = 0; i < 10; ++i) {
    const int Lp = I_LAYER[i];
    // e = silu(LN(e @ W + b))   (edge_conv, WMMA in place)
    gemm64_fused<1, false><<<gE, 256, 0, stream>>>(
        e, nullptr, F(Lp + 0), F(Lp + 1), F(Lp + 3), F(Lp + 2), nullptr, e, NE);

    if (i % 2 == 0) {
      // GINEConv
      fill_f32<<<(N * 64 + 255) / 256, 256, 0, stream>>>(nbuf, 0.f, N * 64);
      gine_msg<<<gE, 256, 0, stream>>>(e, F(Lp + 8), F(Lp + 9), h, src, dst, nbuf, NE);
      gemm64_fused<1, true><<<gN, 256, 0, stream>>>(
          h, nbuf, F(Lp + 4), F(Lp + 5), F(Lp + 11), F(Lp + 10), nullptr, ztmp, N);
      gemm64_fused<2, false><<<gN, 256, 0, stream>>>(
          ztmp, nullptr, F(Lp + 6), F(Lp + 7), F(Lp + 13), F(Lp + 12), h, h, N);
      const int Pp = I_POOL[i / 2];
      gemm64_fused<1, false><<<gN, 256, 0, stream>>>(
          h, nullptr, F(Pp + 0), F(Pp + 1), F(Pp + 3), F(Pp + 2), nullptr, h, N);
    } else {
      // TransformerConv (q,k,v,skip on nodes; e-proj on edges)
      gemm64_fused<0, false><<<gN, 256, 0, stream>>>(
          h, nullptr, F(Lp + 11), F(Lp + 12), nullptr, nullptr, nullptr, qb, N);
      gemm64_fused<0, false><<<gN, 256, 0, stream>>>(
          h, nullptr, F(Lp + 9), F(Lp + 10), nullptr, nullptr, nullptr, kb, N);
      gemm64_fused<0, false><<<gN, 256, 0, stream>>>(
          h, nullptr, F(Lp + 15), F(Lp + 16), nullptr, nullptr, nullptr, vb, N);
      gemm64_fused<0, false><<<gN, 256, 0, stream>>>(
          h, nullptr, F(Lp + 13), F(Lp + 14), nullptr, nullptr, nullptr, skipb, N);
      gemm64_fused<0, false><<<gE, 256, 0, stream>>>(
          e, nullptr, F(Lp + 7), F(Lp + 8), nullptr, nullptr, nullptr, ehb, NE);

      fill_u32<<<(N * 8 + 255) / 256, 256, 0, stream>>>(mord, ORD_NEG, N * 8);
      fill_f32<<<(N * 8 + 255) / 256, 256, 0, stream>>>(sden, 0.f, N * 8);
      fill_f32<<<(N * 64 + 255) / 256, 256, 0, stream>>>(nbuf, 0.f, N * 64);

      attn_alpha<<<gEH, 256, 0, stream>>>(qb, kb, ehb, src, dst, alphab, mord, NE);
      attn_exp<<<gEH, 256, 0, stream>>>(alphab, mord, dst, exb, sden, NE);
      attn_agg<<<gEH, 256, 0, stream>>>(vb, ehb, exb, sden, src, dst, nbuf, NE);
      trans_combine<<<(N + 7) / 8, 256, 0, stream>>>(
          nbuf, skipb, F(Lp + 6), F(Lp + 5), F(Lp + 4), h, N);
    }
  }

  // ---- Set2Set x2 ----
  for (int rd = 0; rd < 2; ++rd) {
    const int R = I_RD[rd];
    fill_f32<<<(NB * 64 + 255) / 256, 256, 0, stream>>>(hB, 0.f, NB * 64);
    fill_f32<<<(NB * 64 + 255) / 256, 256, 0, stream>>>(cB, 0.f, NB * 64);
    fill_f32<<<(NB * 128 + 255) / 256, 256, 0, stream>>>(qstar, 0.f, NB * 128);
    for (int st = 0; st < 4; ++st) {
      lstm_step<<<NB, 256, 0, stream>>>(qstar, hB, cB, F(R + 1), F(R + 0),
                                        F(R + 3), F(R + 2));
      fill_u32<<<(NB + 255) / 256, 256, 0, stream>>>(mordB, ORD_NEG, NB);
      fill_f32<<<(NB + 255) / 256, 256, 0, stream>>>(sdenB, 0.f, NB);
      s2s_energy<<<(N + 255) / 256, 256, 0, stream>>>(h, hB, batch, ener, mordB, N);
      s2s_expsum<<<(N + 255) / 256, 256, 0, stream>>>(ener, mordB, batch, exn, sdenB, N);
      s2s_weighted<<<(N + 255) / 256, 256, 0, stream>>>(h, exn, sdenB, batch, qstar, N);
    }
    copy_strided<<<(NB * 128 + 255) / 256, 256, 0, stream>>>(
        qstar, 128, r4 + rd * 128, 256, 128, NB);
  }

  // ---- head ----
  small_linear<1><<<NB, 128, 0, stream>>>(r4, 256, nullptr, 0,
      F(I_RP + 0), F(I_RP + 1), F(I_RP + 3), F(I_RP + 2), rproj, 128);
  gfx_kernel<<<1, NB, 0, stream>>>(F(4), F(5), F(6), gfxb, NB);
  small_linear<1><<<NB, 64, 0, stream>>>(gfxb, 2, nullptr, 0,
      F(I_GM + 0), F(I_GM + 1), F(I_GM + 5), F(I_GM + 4), t1, 64);
  small_linear<0><<<NB, 64, 0, stream>>>(t1, 64, nullptr, 0,
      F(I_GM + 2), F(I_GM + 3), nullptr, nullptr, gf, 64);
  small_linear<1><<<NB, 64, 0, stream>>>(rproj, 128, gf, 64,
      F(I_FINAL + 0), F(I_FINAL + 1), F(I_FINAL + 7), F(I_FINAL + 6), z1, 64);
  small_linear<1><<<NB, 64, 0, stream>>>(z1, 64, nullptr, 0,
      F(I_FINAL + 2), F(I_FINAL + 3), F(I_FINAL + 9), F(I_FINAL + 8), z2, 32);
  final_out<<<1, NB, 0, stream>>>(z2, F(I_FINAL + 4), F(I_FINAL + 5),
                                  (float*)d_out, NB);
}
